// GAT_41644002902639
// MI455X (gfx1250) — matrix operand order repaired and checked by reading, not compiled, not run on hardware
//
#include <hip/hip_runtime.h>

typedef __attribute__((ext_vector_type(16))) _Float16 v16h;
typedef __attribute__((ext_vector_type(8)))  float    v8f;

#define NN        51200
#define EE        1638400
#define ET        (EE + NN)      // edges incl. self-loops = 1689600
#define HC        64
#define HIDC      32
#define NG        320
#define PER_G     (NN / NG)      // 160
#define NEG_SLOPE 0.2f

// ---- monotone float <-> uint encoding for atomic segment-max ----
__device__ __forceinline__ unsigned fenc(float f) {
    unsigned u = __float_as_uint(f);
    return (u & 0x80000000u) ? ~u : (u | 0x80000000u);
}
__device__ __forceinline__ float fdec(unsigned k) {
    unsigned u = (k & 0x80000000u) ? (k & 0x7FFFFFFFu) : ~k;
    return __uint_as_float(u);
}

// ---------------- zero fill, float4-wide (graph-capture safe init) ----------------
__global__ void gat_zero4(float4* __restrict__ p, int n4) {
    int gid = blockIdx.x * blockDim.x + threadIdx.x;
    if (gid < n4) p[gid] = make_float4(0.f, 0.f, 0.f, 0.f);
}

// ---------------- layer 0 transform: h = x(N,5) @ W0(5,64) ----------------
__global__ void gat_lin0(const float* __restrict__ x, const float* __restrict__ W,
                         float* __restrict__ h) {
    int gid  = blockIdx.x * blockDim.x + threadIdx.x;   // N*64 threads
    int node = gid >> 6, col = gid & 63;
    const float* xr = x + node * 5;
    float acc = 0.0f;
#pragma unroll
    for (int k = 0; k < 5; ++k) acc = fmaf(xr[k], W[k * 64 + col], acc);
    h[gid] = acc;
}

// pack 8 contiguous floats (2 x float4) into halves [i0..i0+7] of a v16h
__device__ __forceinline__ void pack8(v16h& a, int i0, float4 f0, float4 f1) {
    a[i0 + 0] = (_Float16)f0.x; a[i0 + 1] = (_Float16)f0.y;
    a[i0 + 2] = (_Float16)f0.z; a[i0 + 3] = (_Float16)f0.w;
    a[i0 + 4] = (_Float16)f1.x; a[i0 + 5] = (_Float16)f1.y;
    a[i0 + 6] = (_Float16)f1.z; a[i0 + 7] = (_Float16)f1.w;
}

// ---------------- layers 1/2 transform: h = x(N,64) @ W(64,64) via WMMA ----------------
// One wave32 computes a 16x64 output slab: 4 n-tiles x (2 x v_wmma_f32_16x16x32_f16).
// W is staged TRANSPOSED to LDS so every B fragment is 16 contiguous halves
// (one aligned 32B read -> 2x ds_load_b128 instead of 32x ds_load_u16).
__global__ void __launch_bounds__(256) gat_lin_wmma(const float* __restrict__ x,
                                                    const float* __restrict__ W,
                                                    float* __restrict__ h) {
    __shared__ __align__(32) _Float16 sWt[64 * 64];   // sWt[n*64 + k] = W[k*64 + n]
    int t = threadIdx.x;
#pragma unroll
    for (int i = 0; i < 16; ++i) {
        int idx = t + i * 256;
        int k = idx >> 6, n = idx & 63;
        sWt[n * 64 + k] = (_Float16)W[idx];
    }
    __syncthreads();

    int lane    = t & 31;
    int wave    = t >> 5;
    int m0      = (blockIdx.x * 8 + wave) * 16;  // 16-row tile
    int halfSel = lane >> 4;                     // lane group 0/1
    int lrow    = lane & 15;

    // A fragments, contiguous k order: k = 16*halfSel + i
    // -> runs of 16 contiguous floats: load as float4 quads.
    const float4* xr4 = (const float4*)(x + (m0 + lrow) * 64);  // 16 float4 per row
    int kb4 = halfSel << 2;                                     // (16*halfSel)/4
    v16h a0, a1;
    pack8(a0, 0, xr4[kb4 + 0],  xr4[kb4 + 1]);   // K =      kb .. kb+7
    pack8(a0, 8, xr4[kb4 + 2],  xr4[kb4 + 3]);   // K =  8 + kb .. kb+7
    pack8(a1, 0, xr4[kb4 + 8],  xr4[kb4 + 9]);   // K = 32 + kb .. kb+7
    pack8(a1, 8, xr4[kb4 + 10], xr4[kb4 + 11]);  // K = 40 + kb .. kb+7

    int kbB = halfSel << 4;                      // B 32x16: lanes 0-15 K=0..15, 16-31 K=16..31
#pragma unroll
    for (int nt = 0; nt < 4; ++nt) {
        int col = nt * 16 + lrow;
        // 16 contiguous halves, 32B aligned
        v16h b0 = *(const v16h*)&sWt[col * 64 + kbB];        // K = kbB .. kbB+15
        v16h b1 = *(const v16h*)&sWt[col * 64 + 32 + kbB];   // K = 32 + kbB .. +15
        v8f c = {};
        c = __builtin_amdgcn_wmma_f32_16x16x32_f16(false, a0, false, b0, (short)0, c, false, false);
        c = __builtin_amdgcn_wmma_f32_16x16x32_f16(false, a1, false, b1, (short)0, c, false, false);
        // C/D layout: VGPR r -> row m0 + r + 8*halfSel, col = nt*16 + (lane&15)
        int rbase = m0 + (halfSel << 3);
#pragma unroll
        for (int r = 0; r < 8; ++r) h[(rbase + r) * 64 + col] = c[r];
    }
}

// ---------------- per-node attention logits ----------------
__global__ void gat_logits(const float* __restrict__ h, const float* __restrict__ a_src,
                           const float* __restrict__ a_dst,
                           float* __restrict__ alS, float* __restrict__ alD) {
    int gid  = blockIdx.x * blockDim.x + threadIdx.x;   // N*2 threads
    int node = gid >> 1, hd = gid & 1;
    const float* hr = h + node * 64 + hd * 32;
    const float* as = a_src + hd * 32;
    const float* ad = a_dst + hd * 32;
    float s = 0.0f, d = 0.0f;
#pragma unroll
    for (int c = 0; c < 32; ++c) {
        s = fmaf(hr[c], as[c], s);
        d = fmaf(hr[c], ad[c], d);
    }
    alS[gid] = s;
    alD[gid] = d;
}

// ---------------- pass 1: segment max over dst ----------------
__global__ void gat_edge_max(const int* __restrict__ ei, const float* __restrict__ alS,
                             const float* __restrict__ alD, unsigned* __restrict__ mEnc) {
    int gid = blockIdx.x * blockDim.x + threadIdx.x;    // ET*2 threads
    int e = gid >> 1, hd = gid & 1;
    if (e + 8192 < EE) {                                // stream-ahead into GL2
        __builtin_prefetch(&ei[e + 8192], 0, 1);
        __builtin_prefetch(&ei[EE + e + 8192], 0, 1);
    }
    int s, d;
    if (e < EE) { s = ei[e]; d = ei[EE + e]; } else { s = d = e - EE; }
    float v = alS[s * 2 + hd] + alD[d * 2 + hd];
    v = v > 0.0f ? v : NEG_SLOPE * v;
    atomicMax(&mEnc[d * 2 + hd], fenc(v));
}

// ---------------- pass 2: segment sum of exp(e - m) ----------------
__global__ void gat_edge_sum(const int* __restrict__ ei, const float* __restrict__ alS,
                             const float* __restrict__ alD, const unsigned* __restrict__ mEnc,
                             float* __restrict__ denom) {
    int gid = blockIdx.x * blockDim.x + threadIdx.x;    // ET*2 threads
    int e = gid >> 1, hd = gid & 1;
    int s, d;
    if (e < EE) { s = ei[e]; d = ei[EE + e]; } else { s = d = e - EE; }
    float v = alS[s * 2 + hd] + alD[d * 2 + hd];
    v = v > 0.0f ? v : NEG_SLOPE * v;
    float ex = __expf(v - fdec(mEnc[d * 2 + hd]));
    atomicAdd(&denom[d * 2 + hd], ex);
}

// ---------------- pass 3: out[dst] += alpha * h[src]  (wave32 per (edge,head)) ----------------
__global__ void __launch_bounds__(256) gat_edge_aggr(const int* __restrict__ ei,
                                                     const float* __restrict__ alS,
                                                     const float* __restrict__ alD,
                                                     const unsigned* __restrict__ mEnc,
                                                     const float* __restrict__ denom,
                                                     const float* __restrict__ h,
                                                     float* __restrict__ out) {
    int gid  = blockIdx.x * 256 + threadIdx.x;          // ET*2*32 threads
    int wid  = gid >> 5;                                // (edge, head) per wave
    int lane = threadIdx.x & 31;
    int e = wid >> 1, hd = wid & 1;
    int s, d;
    if (e < EE) { s = ei[e]; d = ei[EE + e]; } else { s = d = e - EE; }
    float alpha = 0.0f;
    if (lane == 0) {
        float v = alS[s * 2 + hd] + alD[d * 2 + hd];
        v = v > 0.0f ? v : NEG_SLOPE * v;
        float ex = __expf(v - fdec(mEnc[d * 2 + hd]));
        alpha = ex / denom[d * 2 + hd];
    }
    alpha = __shfl(alpha, 0, 32);                       // wave32 broadcast
    float val = alpha * h[s * 64 + hd * 32 + lane];     // coalesced 128B gather (L2-resident)
    atomicAdd(&out[d * 64 + hd * 32 + lane], val);      // coalesced 128B L2 atomic add
}

// ---------------- x_next = ELU(agg + bias), float4-wide, in place ----------------
__global__ void gat_finalize(float4* __restrict__ agg, const float4* __restrict__ b4) {
    int gid = blockIdx.x * blockDim.x + threadIdx.x;    // N*16 threads
    float4 v = agg[gid];
    float4 b = b4[gid & 15];
    v.x += b.x; v.y += b.y; v.z += b.z; v.w += b.w;
    v.x = v.x > 0.f ? v.x : (__expf(v.x) - 1.f);
    v.y = v.y > 0.f ? v.y : (__expf(v.y) - 1.f);
    v.z = v.z > 0.f ? v.z : (__expf(v.z) - 1.f);
    v.w = v.w > 0.f ? v.w : (__expf(v.w) - 1.f);
    agg[gid] = v;
}

// ---------------- global mean pool (batch is contiguous repeat of 160) ----------------
__global__ void gat_pool(const float* __restrict__ x, float* __restrict__ g) {
    int gid = blockIdx.x * blockDim.x + threadIdx.x;    // NG*64
    int gr = gid >> 6, f = gid & 63;
    const float* p = x + gr * PER_G * 64 + f;
    float acc = 0.0f;
    for (int i = 0; i < PER_G; ++i) acc += p[i * 64];
    g[gid] = acc * (1.0f / PER_G);
}

// ---------------- MLP head: relu(g@Wm1+bm1)@Wm2+bm2 ----------------
__global__ void gat_mlp(const float* __restrict__ g, const float* __restrict__ Wm1,
                        const float* __restrict__ bm1, const float* __restrict__ Wm2,
                        const float* __restrict__ bm2, float* __restrict__ out) {
    int gr = blockIdx.x * blockDim.x + threadIdx.x;
    if (gr >= NG) return;
    const float* gp = g + gr * 64;
    float hid[32];
#pragma unroll
    for (int j = 0; j < 32; ++j) {
        float acc = bm1[j];
        for (int k = 0; k < 64; ++k) acc = fmaf(gp[k], Wm1[k * 32 + j], acc);
        hid[j] = fmaxf(acc, 0.0f);
    }
#pragma unroll
    for (int o = 0; o < 3; ++o) {
        float acc = bm2[o];
        for (int j = 0; j < 32; ++j) acc = fmaf(hid[j], Wm2[j * 3 + o], acc);
        out[gr * 3 + o] = acc;
    }
}

extern "C" void kernel_launch(void* const* d_in, const int* in_sizes, int n_in,
                              void* d_out, int out_size, void* d_ws, size_t ws_size,
                              hipStream_t stream) {
    (void)in_sizes; (void)n_in; (void)out_size; (void)ws_size;

    const float* x_in = (const float*)d_in[0];
    const int*   ei   = (const int*)d_in[1];
    // d_in[2] = batch: contiguous repeat -> graph id derived arithmetically
    const float* W[3]  = {(const float*)d_in[3],  (const float*)d_in[7],  (const float*)d_in[11]};
    const float* aS[3] = {(const float*)d_in[4],  (const float*)d_in[8],  (const float*)d_in[12]};
    const float* aD[3] = {(const float*)d_in[5],  (const float*)d_in[9],  (const float*)d_in[13]};
    const float* bb[3] = {(const float*)d_in[6],  (const float*)d_in[10], (const float*)d_in[14]};
    const float* Wm1 = (const float*)d_in[15];
    const float* bm1 = (const float*)d_in[16];
    const float* Wm2 = (const float*)d_in[17];
    const float* bm2 = (const float*)d_in[18];
    float* out = (float*)d_out;

    // workspace layout
    float*    bufA  = (float*)d_ws;              // N*64
    float*    bufB  = bufA + NN * 64;            // N*64
    float*    bufH  = bufB + NN * 64;            // N*64
    float*    alS   = bufH + NN * 64;            // N*2
    float*    alD   = alS + NN * 2;              // N*2
    unsigned* mEnc  = (unsigned*)(alD + NN * 2); // N*2 (encoded max), contiguous with denom
    float*    denom = (float*)(mEnc + NN * 2);   // N*2
    float*    gbuf  = denom + NN * 2;            // NG*64

    const int B = 256;
    float* aggOf[3] = {bufA, bufB, bufA};        // ping-pong; h always in bufH
    const float* xcur = nullptr;

    for (int l = 0; l < 3; ++l) {
        if (l == 0)
            gat_lin0<<<NN * 64 / B, B, 0, stream>>>(x_in, W[0], bufH);
        else
            gat_lin_wmma<<<NN / 16 / 8, B, 0, stream>>>(xcur, W[l], bufH);

        gat_logits<<<NN * 2 / B, B, 0, stream>>>(bufH, aS[l], aD[l], alS, alD);

        float* agg = aggOf[l];
        gat_zero4<<<NN * 64 / 4 / B, B, 0, stream>>>((float4*)agg, NN * 16);
        gat_zero4<<<NN * 4 / 4 / B, B, 0, stream>>>((float4*)mEnc, NN);  // mEnc (0 sorts below fenc(-inf)) + denom

        gat_edge_max <<<ET * 2 / B, B, 0, stream>>>(ei, alS, alD, mEnc);
        gat_edge_sum <<<ET * 2 / B, B, 0, stream>>>(ei, alS, alD, mEnc, denom);
        gat_edge_aggr<<<(long long)ET * 2 * 32 / B, B, 0, stream>>>(ei, alS, alD, mEnc, denom, bufH, agg);

        gat_finalize<<<NN * 16 / B, B, 0, stream>>>((float4*)agg, (const float4*)bb[l]);
        xcur = agg;
    }

    gat_pool<<<NG * 64 / B, B, 0, stream>>>(xcur, gbuf);
    gat_mlp<<<3, 128, 0, stream>>>(gbuf, Wm1, bm1, Wm2, bm2, out);
}
